// MHA_17042430230986
// MI455X (gfx1250) — matrix operand order repaired and checked by reading, not compiled, not run on hardware
//
#include <hip/hip_runtime.h>
#include <hip/hip_bf16.h>

// ---------------------------------------------------------------------------
// MHA (no softmax) forward for MI455X / gfx1250, bf16 WMMA pipeline.
// B=8 S=1024 D=2048 H=16 HD=128 ROPE=64 start_pos=0
// ---------------------------------------------------------------------------

typedef __attribute__((ext_vector_type(16))) __bf16 v16bf;
typedef __attribute__((ext_vector_type(8)))  __bf16 v8bf;
typedef __attribute__((ext_vector_type(8)))  float  v8f;
typedef int i4v __attribute__((vector_size(16)));   // matches builtin proto

#define B_   8
#define S_   1024
#define D_   2048
#define H_   16
#define HD_  128

#ifndef __has_builtin
#define __has_builtin(x) 0
#endif
#if __has_builtin(__builtin_amdgcn_global_load_async_to_lds_b128) && \
    __has_builtin(__builtin_amdgcn_s_wait_asynccnt)
#define USE_ASYNC 1
#else
#define USE_ASYNC 0
#endif

#if USE_ASYNC
// cast helpers: generic -> addrspace-qualified i4v*
#define AS1_I4V(p) \
    ((__attribute__((address_space(1))) i4v*)(__attribute__((address_space(1))) void*)(p))
#define AS3_I4V(p) \
    ((__attribute__((address_space(3))) i4v*)(__attribute__((address_space(3))) void*)(p))
#endif

// ---------------------------------------------------------------------------
// WMMA fragment loads from pre-resolved per-lane pointers.
// A (16x32 bf16): lane holds row (lane&15); 16 consecutive K values start
// at +0 / +16 (lane<16 / >=16). Two 16B ds/global loads.
// B (32x16 bf16): lane holds column (lane&15); 16 consecutive K values start
// at +0 / +16 (lane<16 / >=16). Two 16B loads, 32B contiguous.
// ---------------------------------------------------------------------------
static __device__ __forceinline__ v16bf frag_ld_a(const __bf16* p) {
    v16bf f;
    ((v8bf*)&f)[0] = *(const v8bf*)(p);
    ((v8bf*)&f)[1] = *(const v8bf*)(p + 8);
    return f;
}
static __device__ __forceinline__ v16bf frag_ld_b(const __bf16* p) {
    v16bf f;
    ((v8bf*)&f)[0] = *(const v8bf*)(p);
    ((v8bf*)&f)[1] = *(const v8bf*)(p + 8);
    return f;
}
static __device__ __forceinline__ v8f wmma_bf16(v16bf a, v16bf b, v8f c) {
    return __builtin_amdgcn_wmma_f32_16x16x32_bf16(false, a, false, b, (short)0, c,
                                                   false, false);
}

// ---------------------------------------------------------------------------
// fp32 -> bf16 conversion (vectorized, n % 4 == 0)
// ---------------------------------------------------------------------------
__global__ void k_cvt_bf16(const float* __restrict__ src, __bf16* __restrict__ dst,
                           int n) {
    int i = (blockIdx.x * blockDim.x + threadIdx.x) * 4;
    if (i + 3 < n) {
        float4 v = *(const float4*)(src + i);
        __bf16 o[4];
        o[0] = (__bf16)v.x; o[1] = (__bf16)v.y;
        o[2] = (__bf16)v.z; o[3] = (__bf16)v.w;
        *(uint2*)(dst + i) = *(uint2*)o;
    }
}

// ---------------------------------------------------------------------------
// C[M,N] = A[M,K] @ B[N,K]^T + bias[N]   (N, K compile-time)
// Block tile 128x128, 256 threads = 8 wave32 waves (4 M-slabs x 2 N-slabs),
// each wave computes 32x64 = 2x4 WMMA tiles of 16x16.
// ---------------------------------------------------------------------------
template <bool OUT_BF16, int N, int K>
__global__ void __launch_bounds__(256)
k_gemm_bias(const __bf16* __restrict__ A, const __bf16* __restrict__ Bw,
            const float* __restrict__ bias, void* __restrict__ Cout) {
    __shared__ __bf16 Asm[128 * 32];
    __shared__ __bf16 Bsm[128 * 32];

    const int tid  = threadIdx.x;
    const int lane = tid & 31;
    const int w    = tid >> 5;
    const int wm   = w & 3;
    const int wn   = w >> 2;
    const int m0   = blockIdx.y * 128;
    const int n0   = blockIdx.x * 128;

    const int ncol   = lane & 15;
    const int rbase  = (lane < 16) ? 0 : 8;
    const int koff_a = (lane < 16) ? 0 : 16;
    const int koff_b = (lane < 16) ? 0 : 16;

    v8f zero = {};
    v8f acc[2][4];
#pragma unroll
    for (int i = 0; i < 2; ++i)
#pragma unroll
        for (int j = 0; j < 4; ++j) acc[i][j] = zero;

    // cooperative staging: each thread owns 16 bf16 (two 16B chunks) per tile
    const int lrow = tid >> 1;
    const int lk   = (tid & 1) * 16;
    const __bf16* gA = A  + (size_t)(m0 + lrow) * K + lk;
    const __bf16* gB = Bw + (size_t)(n0 + lrow) * K + lk;
    __bf16* sA = Asm + lrow * 32 + lk;
    __bf16* sB = Bsm + lrow * 32 + lk;

    // pre-resolved per-lane fragment pointers (immediate ds offsets after this)
    const __bf16* pa = Asm + (wm * 32 + ncol) * 32 + koff_a;
    const __bf16* pb = Bsm + (wn * 64 + ncol) * 32 + koff_b;

    for (int kb = 0; kb < K; kb += 32) {
        if (kb + 32 < K) {                     // global_prefetch_b8 next chunk
            __builtin_prefetch(gA + kb + 32, 0, 1);
            __builtin_prefetch(gB + kb + 32, 0, 1);
        }
        v8bf a0 = *(const v8bf*)(gA + kb);
        v8bf a1 = *(const v8bf*)(gA + kb + 8);
        v8bf b0 = *(const v8bf*)(gB + kb);
        v8bf b1 = *(const v8bf*)(gB + kb + 8);
        __syncthreads();
        *(v8bf*)(sA)     = a0;
        *(v8bf*)(sA + 8) = a1;
        *(v8bf*)(sB)     = b0;
        *(v8bf*)(sB + 8) = b1;
        __syncthreads();

        v16bf af0 = frag_ld_a(pa);
        v16bf af1 = frag_ld_a(pa + 16 * 32);
#pragma unroll
        for (int j = 0; j < 4; ++j) {
            v16bf bf = frag_ld_b(pb + j * 16 * 32);
            acc[0][j] = wmma_bf16(af0, bf, acc[0][j]);
            acc[1][j] = wmma_bf16(af1, bf, acc[1][j]);
        }
    }

    // epilogue: one base address, all stores at immediate offsets
    float bvv[4];
#pragma unroll
    for (int j = 0; j < 4; ++j) bvv[j] = bias[n0 + wn * 64 + j * 16 + ncol];

    const size_t cbase =
        (size_t)(m0 + wm * 32 + rbase) * N + (n0 + wn * 64 + ncol);
    if (OUT_BF16) {
        __bf16* p = (__bf16*)Cout + cbase;
#pragma unroll
        for (int i = 0; i < 2; ++i)
#pragma unroll
            for (int r = 0; r < 8; ++r)
#pragma unroll
                for (int j = 0; j < 4; ++j)
                    p[(size_t)(i * 16 + r) * N + j * 16] =
                        (__bf16)(acc[i][j][r] + bvv[j]);
    } else {
        float* p = (float*)Cout + cbase;
#pragma unroll
        for (int i = 0; i < 2; ++i)
#pragma unroll
            for (int r = 0; r < 8; ++r)
#pragma unroll
                for (int j = 0; j < 4; ++j)
                    p[(size_t)(i * 16 + r) * N + j * 16] =
                        acc[i][j][r] + bvv[j];
    }
}

// ---------------------------------------------------------------------------
// RoPE on first 64 dims of each head of Q and K (in place, bf16 storage).
// ---------------------------------------------------------------------------
__global__ void k_rope(__bf16* __restrict__ Q, __bf16* __restrict__ Kc,
                       const float* __restrict__ cosh,
                       const float* __restrict__ sinh) {
    const int idx = blockIdx.x * blockDim.x + threadIdx.x;
    const int d = idx & 31;
    const int h = (idx >> 5) & 15;
    const int s = (idx >> 9) & 1023;
    const int b = idx >> 19;
    if (b >= B_) return;
    const size_t base = (size_t)(b * S_ + s) * D_ + h * HD_;
    const float c  = cosh[s * 32 + d];
    const float si = sinh[s * 32 + d];
    {
        float x1 = (float)Q[base + d], x2 = (float)Q[base + d + 32];
        Q[base + d]      = (__bf16)(x1 * c - x2 * si);
        Q[base + d + 32] = (__bf16)(x2 * c + x1 * si);
    }
    {
        float x1 = (float)Kc[base + d], x2 = (float)Kc[base + d + 32];
        Kc[base + d]      = (__bf16)(x1 * c - x2 * si);
        Kc[base + d + 32] = (__bf16)(x2 * c + x1 * si);
    }
}

// ---------------------------------------------------------------------------
// Attention core (no softmax): out_head = (Q K^T + mask) V, per (b,h).
// One block per (b, h, 128-row s-tile); 8 waves, 16 s-rows each.
// Q fragments in registers; K tile [t][d] (async-to-LDS when available) and
// transposed V tile [d][t] in LDS; scores bounce via per-wave LDS scratch.
// ---------------------------------------------------------------------------
__global__ void __launch_bounds__(256)
k_attn(const __bf16* __restrict__ Q, const __bf16* __restrict__ Kc,
       const __bf16* __restrict__ V, const float* __restrict__ mask,
       __bf16* __restrict__ Aout) {
    __shared__ __bf16 Ks[128 * 128];      // [t][d]
    __shared__ __bf16 Vt[128 * 128];      // [d][t]  (transposed)
    __shared__ __bf16 Ssc[8][16 * 32];    // per-wave score scratch [s][t]

    const int tid  = threadIdx.x;
    const int lane = tid & 31;
    const int wv   = tid >> 5;
    const int sb   = blockIdx.x & 7;
    const int h    = (blockIdx.x >> 3) & 15;
    const int b    = blockIdx.x >> 7;
    const int s0   = sb * 128;

    const int ncol   = lane & 15;
    const int rbase  = (lane < 16) ? 0 : 8;
    const int koff_a = (lane < 16) ? 0 : 16;
    const int koff_b = (lane < 16) ? 0 : 16;

    // Q fragments: wave's 16 rows, HD=128 -> 4 K-chunks of 32
    v16bf qf[4];
    {
        const __bf16* qp = Q + (size_t)(b * S_ + s0 + wv * 16 + ncol) * D_ +
                           h * HD_ + koff_a;
#pragma unroll
        for (int dk = 0; dk < 4; ++dk) qf[dk] = frag_ld_a(qp + dk * 32);
    }

    v8f zero = {};
    v8f acco[8];
#pragma unroll
    for (int j = 0; j < 8; ++j) acco[j] = zero;

    // per-lane bases: fragment pointers (immediate ds offsets from here)
    const __bf16* Ksl = Ks + ncol * 128 + koff_b;       // scores B operand
    const __bf16* Vtl = Vt + ncol * 128 + koff_b;       // out B operand
    __bf16*       sc  = &Ssc[wv][0];
    const __bf16* scl = sc + ncol * 32 + koff_a;        // scores A operand
    // mask: one 64-bit base, everything else immediate offsets
    const float* maskp =
        mask + (size_t)(s0 + wv * 16 + rbase) * S_ + ncol;

    // cooperative-staging coordinates
    const int trow = tid >> 1;
    const int dh   = (tid & 1) * 64;
    const __bf16* gK = Kc + (size_t)(b * S_ + trow) * D_ + h * HD_ + dh;
    const __bf16* gV = V  + (size_t)(b * S_ + trow) * D_ + h * HD_ + dh;

    for (int tb = 0; tb < S_; tb += 128) {
        const size_t goff = (size_t)tb * D_;
        __syncthreads();                     // prev tile consumers done
#if USE_ASYNC
#pragma unroll
        for (int c = 0; c < 8; ++c) {
            __builtin_amdgcn_global_load_async_to_lds_b128(
                AS1_I4V(gK + goff + c * 8),
                AS3_I4V(Ks + trow * 128 + dh + c * 8),
                0, 0);
        }
#else
#pragma unroll
        for (int c = 0; c < 8; ++c) {
            v8bf kv = *(const v8bf*)(gK + goff + c * 8);
            *(v8bf*)(Ks + trow * 128 + dh + c * 8) = kv;
        }
#endif
#pragma unroll
        for (int c = 0; c < 8; ++c) {
            v8bf vvv = *(const v8bf*)(gV + goff + c * 8);
#pragma unroll
            for (int e = 0; e < 8; ++e)
                Vt[(dh + c * 8 + e) * 128 + trow] = vvv[e];
        }
#if USE_ASYNC
        __builtin_amdgcn_s_wait_asynccnt(0);
#endif
        __syncthreads();

#pragma unroll
        for (int tc = 0; tc < 4; ++tc) {     // 32 t-columns per chunk
            // issue mask loads first (immediate offsets), overlap with WMMA
            float mv[2][8];
#pragma unroll
            for (int jj = 0; jj < 2; ++jj)
#pragma unroll
                for (int r = 0; r < 8; ++r)
                    mv[jj][r] = maskp[(size_t)r * S_ + tb + tc * 32 + jj * 16];

            // scores: 16 s-rows x 32 t-cols, reduced over d (4 chunks of 32)
            v8f sacc[2];
            sacc[0] = zero; sacc[1] = zero;
#pragma unroll
            for (int jj = 0; jj < 2; ++jj)
#pragma unroll
                for (int dk = 0; dk < 4; ++dk) {
                    v16bf kb =
                        frag_ld_b(Ksl + (tc * 32 + jj * 16) * 128 + dk * 32);
                    sacc[jj] = wmma_bf16(qf[dk], kb, sacc[jj]);
                }

            // mask add (fp32) + cvt + relayout via per-wave LDS scratch
#pragma unroll
            for (int jj = 0; jj < 2; ++jj)
#pragma unroll
                for (int r = 0; r < 8; ++r)
                    sc[(rbase + r) * 32 + jj * 16 + ncol] =
                        (__bf16)(sacc[jj][r] + mv[jj][r]);

            // scores(16x32) @ V(32x128): B operand t-contiguous in Vt[d][t]
            v16bf af = frag_ld_a(scl);
#pragma unroll
            for (int j = 0; j < 8; ++j) {
                v16bf vb = frag_ld_b(Vtl + (j * 16) * 128 + tc * 32);
                acco[j] = wmma_bf16(af, vb, acco[j]);
            }
        }
    }

    // store attention output (bf16): one base, immediate offsets
    __bf16* op = Aout + (size_t)(b * S_ + s0 + wv * 16 + rbase) * D_ +
                 h * HD_ + ncol;
#pragma unroll
    for (int r = 0; r < 8; ++r)
#pragma unroll
        for (int j = 0; j < 8; ++j)
            op[(size_t)r * D_ + j * 16] = (__bf16)acco[j][r];
}

// ---------------------------------------------------------------------------
// Host side
// ---------------------------------------------------------------------------
extern "C" void kernel_launch(void* const* d_in, const int* in_sizes, int n_in,
                              void* d_out, int out_size, void* d_ws,
                              size_t ws_size, hipStream_t stream) {
    const float* x     = (const float*)d_in[0];
    const float* wq    = (const float*)d_in[1];
    const float* bq    = (const float*)d_in[2];
    const float* wk    = (const float*)d_in[3];
    const float* bk    = (const float*)d_in[4];
    const float* wv    = (const float*)d_in[5];
    const float* bv    = (const float*)d_in[6];
    const float* wo    = (const float*)d_in[7];
    const float* bo    = (const float*)d_in[8];
    const float* cosh  = (const float*)d_in[9];
    const float* sinh  = (const float*)d_in[10];
    const float* mask  = (const float*)d_in[11];
    float* out = (float*)d_out;

    const size_t XN = (size_t)B_ * S_ * D_;   // 16,777,216
    const size_t WN = (size_t)D_ * D_;        //  4,194,304

    __bf16* xbf  = (__bf16*)d_ws;
    __bf16* wqbf = xbf  + XN;
    __bf16* wkbf = wqbf + WN;
    __bf16* wvbf = wkbf + WN;
    __bf16* wobf = wvbf + WN;
    __bf16* Qbf  = wobf + WN;
    __bf16* Kbf  = Qbf  + XN;
    __bf16* Vbf  = Kbf  + XN;
    __bf16* Abf  = xbf;                       // x dead after projections

    const int M = B_ * S_;                    // 8192

    // 1) fp32 -> bf16
    k_cvt_bf16<<<(int)(XN / 4 / 256), 256, 0, stream>>>(x,  xbf,  (int)XN);
    k_cvt_bf16<<<(int)(WN / 4 / 256), 256, 0, stream>>>(wq, wqbf, (int)WN);
    k_cvt_bf16<<<(int)(WN / 4 / 256), 256, 0, stream>>>(wk, wkbf, (int)WN);
    k_cvt_bf16<<<(int)(WN / 4 / 256), 256, 0, stream>>>(wv, wvbf, (int)WN);
    k_cvt_bf16<<<(int)(WN / 4 / 256), 256, 0, stream>>>(wo, wobf, (int)WN);

    // 2) Q/K/V projections (bf16 out)
    dim3 ggrid(D_ / 128, M / 128);
    k_gemm_bias<true, D_, D_><<<ggrid, 256, 0, stream>>>(xbf, wqbf, bq, Qbf);
    k_gemm_bias<true, D_, D_><<<ggrid, 256, 0, stream>>>(xbf, wkbf, bk, Kbf);
    k_gemm_bias<true, D_, D_><<<ggrid, 256, 0, stream>>>(xbf, wvbf, bv, Vbf);

    // 3) RoPE on Q and K
    k_rope<<<(B_ * S_ * H_ * 32) / 256, 256, 0, stream>>>(Qbf, Kbf, cosh, sinh);

    // 4) attention core (no softmax): (QK^T + mask) V
    k_attn<<<B_ * H_ * (S_ / 128), 256, 0, stream>>>(Qbf, Kbf, Vbf, mask, Abf);

    // 5) output projection (fp32 out)
    k_gemm_bias<false, D_, D_><<<ggrid, 256, 0, stream>>>(Abf, wobf, bo, out);
}
